// CausalSelfAttention_68496138437292
// MI455X (gfx1250) — matrix operand order repaired and checked by reading, not compiled, not run on hardware
//
#include <hip/hip_runtime.h>

#define DIM 1024
#define NH 16
#define HD 64
#define S_LEN 2048
#define BATCH 2
#define NTOK (BATCH * S_LEN)   // 4096
#define QKV_ROWS 3072
#define FEAT 16
#define OUT_O 1024

typedef __attribute__((ext_vector_type(16))) __bf16 v16bf;
typedef __attribute__((ext_vector_type(8)))  float  v8f;
typedef __attribute__((ext_vector_type(4)))  unsigned int v4u;

union Frag16 {
  v16bf bf;
  v4u   q[2];
  unsigned short u[16];
};

__device__ __forceinline__ unsigned short f2bf(float f) {
  union { float f; unsigned int u; } v; v.f = f;
  unsigned int r = v.u + 0x7FFFu + ((v.u >> 16) & 1u);   // RNE
  return (unsigned short)(r >> 16);
}
__device__ __forceinline__ float bf2f(unsigned short h) {
  union { unsigned int u; float f; } v; v.u = ((unsigned int)h) << 16;
  return v.f;
}
__device__ __forceinline__ float redSum32(float v) {
#pragma unroll
  for (int m = 1; m < 32; m <<= 1) v += __shfl_xor(v, m, 32);
  return v;
}
__device__ __forceinline__ float redSum16(float v) {
#pragma unroll
  for (int m = 1; m < 16; m <<= 1) v += __shfl_xor(v, m, 32);
  return v;
}
__device__ __forceinline__ float redMax16(float v) {
#pragma unroll
  for (int m = 1; m < 16; m <<= 1) v = fmaxf(v, __shfl_xor(v, m, 32));
  return v;
}

// ---------------- K1: ternary-quantize w_qkv -> bf16 (wave per 64-group) ----
__global__ __launch_bounds__(256) void k_quant_w(const float* __restrict__ w,
                                                 unsigned short* __restrict__ wt) {
  int wid = threadIdx.x >> 5, lane = threadIdx.x & 31;
  int gw = blockIdx.x * 8 + wid;            // 49152 groups
  int row = gw >> 4, gcol = (gw & 15) << 6;
  int base = row * DIM + gcol;
  float w0 = bf2f(f2bf(w[base + lane]));
  float w1 = bf2f(f2bf(w[base + 32 + lane]));
  float s = fmaxf(redSum32(fabsf(w0) + fabsf(w1)) * (1.0f / 64.0f), 1e-8f);
  float q0 = fminf(fmaxf(roundf(w0 / s), -1.f), 1.f);
  float q1 = fminf(fmaxf(roundf(w1 / s), -1.f), 1.f);
  wt[base + lane]      = f2bf(q0 * s);
  wt[base + 32 + lane] = f2bf(q1 * s);
}

// ---------------- K1b: x fp32 -> bf16 -------------------------------------
__global__ __launch_bounds__(256) void k_x2bf(const float* __restrict__ x,
                                              unsigned short* __restrict__ xb) {
  int i0 = blockIdx.x * 1024 + threadIdx.x;
#pragma unroll
  for (int k = 0; k < 4; ++k) { int i = i0 + k * 256; xb[i] = f2bf(x[i]); }
}

// ---------------- K2: QKV GEMM via WMMA bf16, 16x64 tile per wave ----------
__global__ __launch_bounds__(256) void k_gemm_qkv(const unsigned short* __restrict__ xb,
                                                  const unsigned short* __restrict__ wt,
                                                  float* __restrict__ qkv) {
  int wid = threadIdx.x >> 5, lane = threadIdx.x & 31;
  int gw = blockIdx.x * 8 + wid;            // 12288 wave-tiles
  int n64 = gw >> 8;                        // 0..47
  int m16 = gw & 255;                       // 0..255
  int m0 = m16 << 4, n0 = n64 << 6;
  int lm = lane & 15, lh = lane >> 4;

  v8f zero = {0, 0, 0, 0, 0, 0, 0, 0};
  v8f acc[4] = {zero, zero, zero, zero};
  const unsigned short* arow = xb + (size_t)(m0 + lm) * DIM;

  for (int k0 = 0; k0 < DIM; k0 += 32) {
    Frag16 a;
    a.q[0] = *(const v4u*)(arow + k0 + lh * 16);
    a.q[1] = *(const v4u*)(arow + k0 + lh * 16 + 8);
#pragma unroll
    for (int j = 0; j < 4; ++j) {
      Frag16 b;
      const unsigned short* brow = wt + (size_t)(n0 + j * 16 + lm) * DIM + k0 + lh * 16;
      b.q[0] = *(const v4u*)(brow);
      b.q[1] = *(const v4u*)(brow + 8);
      acc[j] = __builtin_amdgcn_wmma_f32_16x16x32_bf16(false, a.bf, false, b.bf,
                                                       (short)0, acc[j], false, false);
    }
  }
#pragma unroll
  for (int j = 0; j < 4; ++j)
#pragma unroll
    for (int r = 0; r < 8; ++r)
      qkv[(size_t)(m0 + r + 8 * lh) * QKV_ROWS + n0 + j * 16 + lm] = acc[j][r];
}

// ---------------- K3: rmsnorm + YaRN rope + gain, repack Q/K/Vt ------------
__global__ __launch_bounds__(256) void k_epilogue(const float* __restrict__ qkv,
                                                  const float* __restrict__ q_gain,
                                                  unsigned short* __restrict__ Qm,
                                                  unsigned short* __restrict__ Km,
                                                  unsigned short* __restrict__ Vt) {
  int wid = threadIdx.x >> 5, lane = threadIdx.x & 31;
  int gw = blockIdx.x * 8 + wid;            // 65536 = NTOK*NH
  int t = gw >> 4, h = gw & 15;
  int b = t >> 11, s = t & 2047;
  const float* row = qkv + (size_t)t * QKV_ROWS + h * HD;
  float q0 = row[lane],            q1 = row[lane + 32];
  float k0 = row[DIM + lane],      k1 = row[DIM + lane + 32];
  float v0 = row[2 * DIM + lane],  v1 = row[2 * DIM + lane + 32];
  float rq = rsqrtf(redSum32(q0 * q0 + q1 * q1) * (1.0f / 64.0f) + 1e-5f);
  float rk = rsqrtf(redSum32(k0 * k0 + k1 * k1) * (1.0f / 64.0f) + 1e-5f);
  q0 *= rq; q1 *= rq; k0 *= rk; k1 *= rk;
  // YaRN rope: i = lane (0..31)
  float e = (float)lane * (1.0f / 32.0f);
  float inv = __expf(-e * 9.210340371976184f);          // 10000^-e
  float ramp = fminf(fmaxf((e - 0.25f) * (1.0f / 0.75f), 0.f), 1.f);
  inv /= (ramp * 3.0f + 1.0f);                          // TRAIN/YARN = 1/4
  float ang = (float)s * inv;
  float c = __cosf(ang), sn = __sinf(ang);
  float oq0 =  q0 * c + q1 * sn, oq1 = -q0 * sn + q1 * c;
  float ok0 =  k0 * c + k1 * sn, ok1 = -k0 * sn + k1 * c;
  float g = q_gain[h];
  oq0 *= g; oq1 *= g;
  size_t bh2 = ((size_t)(b * NH + h)) * 2;
  size_t i0 = (bh2 + 0) * S_LEN * 32 + (size_t)s * 32 + lane;
  size_t i1 = (bh2 + 1) * S_LEN * 32 + (size_t)s * 32 + lane;
  Qm[i0] = f2bf(oq0); Qm[i1] = f2bf(oq1);
  Km[i0] = f2bf(ok0); Km[i1] = f2bf(ok1);
  Vt[((bh2 + 0) * 32 + lane) * S_LEN + s] = f2bf(v0);
  Vt[((bh2 + 1) * 32 + lane) * S_LEN + s] = f2bf(v1);
}

// ---------------- K4: flash causal attention, 1 wave per 16-query block ----
__global__ __launch_bounds__(32) void k_attn(const unsigned short* __restrict__ Qm,
                                             const unsigned short* __restrict__ Km,
                                             const unsigned short* __restrict__ Vt,
                                             float* __restrict__ att) {
  __shared__ unsigned short P[16 * 32];
  int lane = threadIdx.x & 31;
  int bid = blockIdx.x;                     // B*NH*2*(S/16) = 8192
  int qb = bid & 127; int rest = bid >> 7;
  int half = rest & 1; int h = (rest >> 1) & 15; int b = rest >> 5;
  int q0 = qb << 4;
  size_t bh2 = ((size_t)(b * NH + h)) * 2 + half;
  const unsigned short* qbase = Qm + bh2 * S_LEN * 32;
  const unsigned short* kbase = Km + bh2 * S_LEN * 32;
  const unsigned short* vbase = Vt + bh2 * 32 * S_LEN;
  float* abase = att + bh2 * S_LEN * 32;

  int lm = lane & 15, lh = lane >> 4;
  Frag16 aq;
  {
    const unsigned short* qrow = qbase + (size_t)(q0 + lm) * 32;
    aq.q[0] = *(const v4u*)(qrow + lh * 16);
    aq.q[1] = *(const v4u*)(qrow + lh * 16 + 8);
  }
  v8f zero = {0, 0, 0, 0, 0, 0, 0, 0};
  v8f o0 = zero, o1 = zero;
  float mrow[8], srow[8];
#pragma unroll
  for (int r = 0; r < 8; ++r) { mrow[r] = -1e30f; srow[r] = 0.f; }
  const float scale = 0.17677669529663687f;   // 1/sqrt(32)

  for (int kb = 0; kb < q0 + 16; kb += 32) {
    Frag16 bk0, bk1;
    {
      const unsigned short* kr0 = kbase + (size_t)(kb + lm) * 32 + lh * 16;
      const unsigned short* kr1 = kbase + (size_t)(kb + 16 + lm) * 32 + lh * 16;
      bk0.q[0] = *(const v4u*)(kr0); bk0.q[1] = *(const v4u*)(kr0 + 8);
      bk1.q[0] = *(const v4u*)(kr1); bk1.q[1] = *(const v4u*)(kr1 + 8);
    }
    v8f s0 = __builtin_amdgcn_wmma_f32_16x16x32_bf16(false, aq.bf, false, bk0.bf,
                                                     (short)0, zero, false, false);
    v8f s1 = __builtin_amdgcn_wmma_f32_16x16x32_bf16(false, aq.bf, false, bk1.bf,
                                                     (short)0, zero, false, false);
    bool needMask = (kb + 31 > q0);
#pragma unroll
    for (int r = 0; r < 8; ++r) {
      int m = r + 8 * lh;
      int qi = q0 + m;
      float v0 = s0[r] * scale;
      float v1 = s1[r] * scale;
      if (needMask) {
        if (kb + lm      > qi) v0 = -1e30f;
        if (kb + 16 + lm > qi) v1 = -1e30f;
      }
      float mx   = redMax16(fmaxf(v0, v1));
      float mnew = fmaxf(mrow[r], mx);
      float corr = __expf(mrow[r] - mnew);
      float p0 = __expf(v0 - mnew);
      float p1 = __expf(v1 - mnew);
      srow[r] = srow[r] * corr + redSum16(p0 + p1);
      mrow[r] = mnew;
      o0[r] *= corr; o1[r] *= corr;
      P[m * 32 + lm]      = f2bf(p0);
      P[m * 32 + 16 + lm] = f2bf(p1);
    }
    __syncthreads();
    Frag16 ap;
    ap.q[0] = *(const v4u*)(&P[lm * 32 + lh * 16]);
    ap.q[1] = *(const v4u*)(&P[lm * 32 + lh * 16 + 8]);
    __syncthreads();
    Frag16 bv0, bv1;
    {
      const unsigned short* vr0 = vbase + (size_t)lm * S_LEN + kb + lh * 16;
      const unsigned short* vr1 = vbase + (size_t)(16 + lm) * S_LEN + kb + lh * 16;
      bv0.q[0] = *(const v4u*)(vr0); bv0.q[1] = *(const v4u*)(vr0 + 8);
      bv1.q[0] = *(const v4u*)(vr1); bv1.q[1] = *(const v4u*)(vr1 + 8);
    }
    o0 = __builtin_amdgcn_wmma_f32_16x16x32_bf16(false, ap.bf, false, bv0.bf,
                                                 (short)0, o0, false, false);
    o1 = __builtin_amdgcn_wmma_f32_16x16x32_bf16(false, ap.bf, false, bv1.bf,
                                                 (short)0, o1, false, false);
  }
#pragma unroll
  for (int r = 0; r < 8; ++r) {
    int m = r + 8 * lh;
    float invs = 1.0f / srow[r];
    abase[(size_t)(q0 + m) * 32 + lm]      = o0[r] * invs;
    abase[(size_t)(q0 + m) * 32 + 16 + lm] = o1[r] * invs;
  }
}

// ---------------- K5: ternary prototypes -> p_a, (1-p_s) tables ------------
__global__ __launch_bounds__(256) void k_proto(const float* __restrict__ prot,
                                               const float* __restrict__ feat,
                                               float* __restrict__ pa,
                                               float* __restrict__ pcomp) {
  int wid = threadIdx.x >> 5, lane = threadIdx.x & 31;
  int o = blockIdx.x * 8 + wid;             // 1024 rows
  float acc[FEAT];
#pragma unroll
  for (int f = 0; f < FEAT; ++f) acc[f] = 0.f;
  for (int g = 0; g < 16; ++g) {
    int base = o * DIM + g * 64;
    float w0 = bf2f(f2bf(prot[base + lane]));
    float w1 = bf2f(f2bf(prot[base + 32 + lane]));
    float sc = fmaxf(redSum32(fabsf(w0) + fabsf(w1)) * (1.0f / 64.0f), 1e-8f);
    float d0 = fminf(fmaxf(roundf(w0 / sc), -1.f), 1.f) * sc;
    float d1 = fminf(fmaxf(roundf(w1 / sc), -1.f), 1.f) * sc;
#pragma unroll
    for (int f = 0; f < FEAT; ++f) {
      const float* fr = feat + f * DIM + g * 64;
      acc[f] += d0 * fr[lane] + d1 * fr[32 + lane];
    }
  }
#pragma unroll
  for (int f = 0; f < FEAT; ++f) acc[f] = redSum32(acc[f]);
  if (lane == 0) {
#pragma unroll
    for (int f = 0; f < FEAT; ++f) {
      float pf = acc[f];
      float ps = 1.0f / (1.0f + __expf(-5.0f * pf));
      pa[o * FEAT + f]    = pf * ps;
      pcomp[o * FEAT + f] = 1.0f - ps;
    }
  }
}

// ---------------- K6: diff-lambda combine + fused Tversky projection -------
__global__ __launch_bounds__(256) void k_final(const float* __restrict__ att,
                                               const float* __restrict__ feat,
                                               const float* __restrict__ pa,
                                               const float* __restrict__ pcomp,
                                               const float* __restrict__ theta,
                                               const float* __restrict__ alpha,
                                               const float* __restrict__ beta,
                                               const float* __restrict__ lam,
                                               float* __restrict__ out) {
  __shared__ float y[DIM];
  __shared__ float red[256];
  __shared__ float xa[FEAT], oxs[FEAT];
  int tid = threadIdx.x;
  int t = blockIdx.x;
  int b = t >> 11, s = t & 2047;
  for (int d = tid; d < DIM; d += 256) {
    int h = d >> 6, j = d & 63;
    float l = lam[h];
    size_t bh2 = ((size_t)(b * NH + h)) * 2;
    int j2 = (j < 32) ? j : (j - 32);
    float y1 = att[(bh2 + 0) * S_LEN * 32 + (size_t)s * 32 + j2];
    float y2 = att[(bh2 + 1) * S_LEN * 32 + (size_t)s * 32 + j2];
    y[d] = (j < 32) ? (y1 - l * y2) : (y1 + l * y2);
  }
  __syncthreads();
  int f = tid >> 4, c = tid & 15;
  float p = 0.f;
  const float* fr = feat + f * DIM + c * 64;
  const float* yr = y + c * 64;
#pragma unroll 8
  for (int e = 0; e < 64; ++e) p += yr[e] * fr[e];
  red[tid] = p;
  __syncthreads();
  if (tid < FEAT) {
    float sum = 0.f;
#pragma unroll
    for (int k = 0; k < 16; ++k) sum += red[tid * 16 + k];
    float xs = 1.0f / (1.0f + __expf(-5.0f * sum));
    xa[tid]  = sum * xs;
    oxs[tid] = 1.0f - xs;
  }
  __syncthreads();
  float th = fabsf(theta[0]), al = fabsf(alpha[0]), be = fabsf(beta[0]);
#pragma unroll
  for (int oo = 0; oo < 4; ++oo) {
    int o = tid + oo * 256;
    float a1 = 0.f, a2 = 0.f, a3 = 0.f;
#pragma unroll
    for (int ff = 0; ff < FEAT; ++ff) {
      float pav = pa[o * FEAT + ff];
      float pcv = pcomp[o * FEAT + ff];
      a1 += xa[ff]  * pav;
      a2 += xa[ff]  * pcv;
      a3 += oxs[ff] * pav;
    }
    out[(size_t)t * DIM + o] = th * a1 - al * a2 - be * a3;
  }
}

// ---------------- launcher -------------------------------------------------
extern "C" void kernel_launch(void* const* d_in, const int* in_sizes, int n_in,
                              void* d_out, int out_size, void* d_ws, size_t ws_size,
                              hipStream_t stream) {
  (void)in_sizes; (void)n_in; (void)out_size; (void)ws_size;
  const float* x        = (const float*)d_in[0];
  const float* w_qkv    = (const float*)d_in[1];
  const float* features = (const float*)d_in[2];
  const float* prot     = (const float*)d_in[3];
  const float* theta    = (const float*)d_in[4];
  const float* alpha    = (const float*)d_in[5];
  const float* beta     = (const float*)d_in[6];
  const float* q_gain   = (const float*)d_in[7];
  const float* lam      = (const float*)d_in[8];
  float* out = (float*)d_out;

  char* ws = (char*)d_ws;
  size_t off = 0;
  auto carve = [&](size_t bytes) {
    char* p = ws + off;
    off += (bytes + 255) & ~(size_t)255;
    return p;
  };
  unsigned short* wt  = (unsigned short*)carve((size_t)QKV_ROWS * DIM * 2);        // 6.3 MB
  unsigned short* xb  = (unsigned short*)carve((size_t)NTOK * DIM * 2);            // 8.4 MB
  float*          qkv = (float*)carve((size_t)NTOK * QKV_ROWS * 4);                // 50 MB
  unsigned short* Qm  = (unsigned short*)carve((size_t)BATCH * NH * 2 * S_LEN * 32 * 2);
  unsigned short* Km  = (unsigned short*)carve((size_t)BATCH * NH * 2 * S_LEN * 32 * 2);
  unsigned short* Vt  = (unsigned short*)carve((size_t)BATCH * NH * 2 * 32 * S_LEN * 2);
  float*          at  = (float*)carve((size_t)BATCH * NH * 2 * S_LEN * 32 * 4);    // 16.8 MB
  float*          pa  = (float*)carve((size_t)OUT_O * FEAT * 4);
  float*          pc  = (float*)carve((size_t)OUT_O * FEAT * 4);

  k_quant_w <<<6144, 256, 0, stream>>>(w_qkv, wt);
  k_x2bf    <<<4096, 256, 0, stream>>>(x, xb);
  k_gemm_qkv<<<1536, 256, 0, stream>>>(xb, wt, qkv);
  k_epilogue<<<8192, 256, 0, stream>>>(qkv, q_gain, Qm, Km, Vt);
  k_attn    <<<8192,  32, 0, stream>>>(Qm, Km, Vt, at);
  k_proto   <<< 128, 256, 0, stream>>>(prot, features, pa, pc);
  k_final   <<<4096, 256, 0, stream>>>(at, features, pa, pc, theta, alpha, beta, lam, out);
}
